// MoESwiGLU_27058293965530
// MI455X (gfx1250) — compile-verified, matrix operand order checked by reading, not run on hardware
//
#include <hip/hip_runtime.h>

typedef __bf16 bf16;
typedef __attribute__((ext_vector_type(16))) __bf16 v16bf;
typedef __attribute__((ext_vector_type(8)))  __bf16 v8bf;
typedef __attribute__((ext_vector_type(2)))  __bf16 bf16x2;
typedef __attribute__((ext_vector_type(8)))  float  v8f;

#define TT    2048
#define DD    1024
#define NN    4
#define EE    16
#define NDF   4096     // n*d
#define NTOK  4096     // B*T
#define DFFN  1656
#define DFFNP 1664     // padded to multiple of 32
#define EPSF  1.1920929e-7f
#define BK    32
#define BKP   40       // LDS row padding (keeps 16B alignment, breaks bank conflicts)

__device__ __forceinline__ float sigf(float x)  { return 1.0f / (1.0f + __expf(-x)); }
__device__ __forceinline__ float siluf(float x) { return x / (1.0f + __expf(-x)); }

// CDNA5 async global->LDS copy, 16 bytes per lane (tracked by ASYNCcnt).
__device__ __forceinline__ void async_ld16(unsigned lds_off, const void* gptr) {
  asm volatile("global_load_async_to_lds_b128 %0, %1, off"
               :: "v"(lds_off), "v"((unsigned long long)gptr)
               : "memory");
}
// Drain everything (used on the last tile).
__device__ __forceinline__ void wait_async0() {
  asm volatile("s_wait_asynccnt 0" ::: "memory");
}
// Double-buffer pipeline wait: each wave issues exactly 3 async copies per
// k-tile, so ASYNCcnt<=3 means "previous tile landed, next tile in flight".
__device__ __forceinline__ void wait_async3() {
  asm volatile("s_wait_asynccnt 3" ::: "memory");
}
__device__ __forceinline__ unsigned lds_off(const void* p) {
  return (unsigned)(unsigned long long)p;   // LDS aperture: addr[31:0] = LDS offset
}

// Build a v16bf WMMA A/B fragment for this lane from an LDS tile row.
// 16-bit operand layout (ISA 7.12.2): lane L -> row L%16, K = 8*(L/16) + {0..7} and {16..23}.
__device__ __forceinline__ v16bf frag_ld(const bf16* p) {
  v8bf lo = *(const v8bf*)(p);
  v8bf hi = *(const v8bf*)(p + 16);
  return __builtin_shufflevector(lo, hi, 0,1,2,3,4,5,6,7,8,9,10,11,12,13,14,15);
}

// ---------------------------------------------------------------------------
// Kernel 1: fused base RMSNorm (over nd=4096) -> bf16, and router softmax/top-k
// ---------------------------------------------------------------------------
__global__ __launch_bounds__(256) void k_base_gate(
    const float* __restrict__ strm, const float* __restrict__ rw,
    bf16* __restrict__ base, float* __restrict__ gate)
{
  const int token = blockIdx.x;
  const int b = token >> 11, t = token & (TT - 1);
  const int tid = threadIdx.x, lane = tid & 31, wave = tid >> 5;
  const size_t sb = ((size_t)(b * NN) * TT + t) * DD;

  float x[NN][4], sm[4], ssq = 0.f;
#pragma unroll
  for (int q = 0; q < 4; ++q) {
    const int dd = tid + q * 256;
    float s = 0.f;
#pragma unroll
    for (int n = 0; n < NN; ++n) {
      float v = strm[sb + (size_t)n * TT * DD + dd];
      x[n][q] = v; ssq += v * v; s += v;
    }
    sm[q] = s * 0.25f;
  }
  float racc[EE];
#pragma unroll
  for (int e = 0; e < EE; ++e) racc[e] = 0.f;
#pragma unroll
  for (int q = 0; q < 4; ++q) {
    const int dd = tid + q * 256;
#pragma unroll
    for (int e = 0; e < EE; ++e) racc[e] += sm[q] * rw[e * DD + dd];
  }
#pragma unroll
  for (int off = 16; off > 0; off >>= 1) {
    ssq += __shfl_xor(ssq, off, 32);
#pragma unroll
    for (int e = 0; e < EE; ++e) racc[e] += __shfl_xor(racc[e], off, 32);
  }
  __shared__ float sh[8][17];
  __shared__ float s_logit[EE];
  __shared__ float s_scale;
  if (lane == 0) {
    sh[wave][0] = ssq;
    for (int e = 0; e < EE; ++e) sh[wave][1 + e] = racc[e];
  }
  __syncthreads();
  if (tid < EE) {
    float a = 0.f;
    for (int w = 0; w < 8; ++w) a += sh[w][1 + tid];
    s_logit[tid] = a;
  } else if (tid == EE) {
    float a = 0.f;
    for (int w = 0; w < 8; ++w) a += sh[w][0];
    s_scale = rsqrtf(a * (1.0f / NDF) + EPSF);
  }
  __syncthreads();

  if (tid == 0) {                            // softmax + top-prob mask (MAX_K=4, TH=0.8)
    float p[EE], mx = s_logit[0];
    for (int e = 1; e < EE; ++e) mx = fmaxf(mx, s_logit[e]);
    float sum = 0.f;
    for (int e = 0; e < EE; ++e) { p[e] = __expf(s_logit[e] - mx); sum += p[e]; }
    float inv = 1.f / sum;
    for (int e = 0; e < EE; ++e) p[e] *= inv;
    int ord[EE];
    for (int e = 0; e < EE; ++e) ord[e] = e;
    for (int i = 1; i < EE; ++i) {           // stable insertion sort, descending
      int o = ord[i]; float pv = p[o]; int j = i - 1;
      while (j >= 0 && p[ord[j]] < pv) { ord[j + 1] = ord[j]; --j; }
      ord[j + 1] = o;
    }
    float outp[EE];
    for (int e = 0; e < EE; ++e) outp[e] = 0.f;
    float cs = 0.f;
    for (int r = 0; r < EE; ++r) {
      int i2 = ord[r];
      if (r == 0 || (cs < 0.8f && r < 4)) outp[i2] = p[i2];
      cs += p[i2];
    }
    for (int e = 0; e < EE; ++e) gate[(size_t)token * EE + e] = outp[e];
  }

  const float sc = s_scale;
#pragma unroll
  for (int q = 0; q < 4; ++q) {
    const int dd = tid + q * 256;
#pragma unroll
    for (int n = 0; n < NN; ++n)
      base[(size_t)token * NDF + n * DD + dd] = (bf16)(x[n][q] * sc);
  }
}

// ---------------------------------------------------------------------------
// Kernel 2: fold mhc_norm_w into H-projection weights (bf16).
// Row order per expert: [0..3]=phi_pre, [4..7]=phi_post, [8..23]=phi_res.
// ---------------------------------------------------------------------------
__global__ __launch_bounds__(256) void k_fold_wh(
    const float* __restrict__ pre, const float* __restrict__ post,
    const float* __restrict__ res, const float* __restrict__ mhcw,
    bf16* __restrict__ WH)
{
  const int r = blockIdx.x;                 // 0..EE*24-1
  const int e = r / 24, j = r % 24;
  const float* src = (j < 4) ? pre  + ((size_t)e * 4  + j)        * NDF
                   : (j < 8) ? post + ((size_t)e * 4  + (j - 4))  * NDF
                             : res  + ((size_t)e * 16 + (j - 8))  * NDF;
  const float* nw = mhcw + (size_t)e * NDF;
  for (int k = threadIdx.x; k < NDF; k += 256)
    WH[(size_t)r * NDF + k] = (bf16)(src[k] * nw[k]);
}

// ---------------------------------------------------------------------------
// Weight fp32 -> bf16 conversion with row/col zero-padding (4 elems/thread,
// lowers to packed v_cvt_pk_bf16_f32).
// ---------------------------------------------------------------------------
__global__ __launch_bounds__(256) void k_cvtw(
    const float* __restrict__ src, bf16* __restrict__ dst,
    int rows_s, int cols_s, int rows_d, int cols_d)
{
  const int idx = blockIdx.x * 256 + threadIdx.x;      // 4-element group
  const int gpr = cols_d >> 2;                         // groups per dst row
  if (idx >= rows_d * gpr) return;
  const int r = idx / gpr, c = (idx % gpr) * 4;
  float4 f = (r < rows_s && c < cols_s)
             ? *(const float4*)&src[(size_t)r * cols_s + c]
             : make_float4(0.f, 0.f, 0.f, 0.f);
  bf16x2 p0 = { (bf16)f.x, (bf16)f.y };
  bf16x2 p1 = { (bf16)f.z, (bf16)f.w };
  uint2 u;
  u.x = __builtin_bit_cast(unsigned, p0);
  u.y = __builtin_bit_cast(unsigned, p1);
  *(uint2*)&dst[(size_t)r * cols_d + c] = u;
}

// ---------------------------------------------------------------------------
// bf16 WMMA GEMM:  C[M,N] = epi(A[M,K] @ B[N,K]^T), A/B bf16, all dims padded.
// Block tile 64x128, 8 waves (4 M x 2 N), wave tile 16x64.
// Double-buffered LDS; tiles filled with async global->LDS b128 copies:
// issue tile k+1, s_wait_asynccnt 3 (drains tile k, leaves k+1 in flight).
// EPI: 0 = fp32 store, 1 = silu->bf16, 2 = sigmoid->bf16, 3 = *gsig -> fp32
// ---------------------------------------------------------------------------
template <int EPI>
__global__ __launch_bounds__(256) void k_gemm(
    const bf16* __restrict__ A, int lda,
    const bf16* __restrict__ B, int ldb,
    void* __restrict__ C, int ldc,
    const bf16* __restrict__ gsig,
    const float* __restrict__ gate, int expert, int K)
{
  __shared__ __align__(16) bf16 As[2][64][BKP];
  __shared__ __align__(16) bf16 Bs[2][128][BKP];
  __shared__ int s_flag;
  const int tid = threadIdx.x;
  const int m0 = blockIdx.x * 64, n0 = blockIdx.y * 128;

  if (gate) {                                // top-k routing: skip all-zero tiles
    if (tid == 0) s_flag = 0;
    __syncthreads();
    if (tid < 64 && gate[(size_t)(m0 + tid) * EE + expert] != 0.f) s_flag = 1;
    __syncthreads();
    if (!s_flag) return;
  }

  const int lane = tid & 31, wave = tid >> 5;
  const int wm = wave >> 1, wn = wave & 1;
  v8f acc[4] = {};

  // per-thread 16B fill granules: A 64x32 (1/thread), B 128x32 (2/thread)
  const int ra = tid >> 2, ca = (tid & 3) * 8;
  const int rb1 = (tid + 256) >> 2;
  const unsigned aoff[2] = { lds_off(&As[0][ra][ca]),  lds_off(&As[1][ra][ca]) };
  const unsigned boff0[2] = { lds_off(&Bs[0][ra][ca]), lds_off(&Bs[1][ra][ca]) };
  const unsigned boff1[2] = { lds_off(&Bs[0][rb1][ca]), lds_off(&Bs[1][rb1][ca]) };

  auto issue = [&](int k0, int buf) {
    async_ld16(aoff[buf],  A + (size_t)(m0 + ra)  * lda + k0 + ca);
    async_ld16(boff0[buf], B + (size_t)(n0 + ra)  * ldb + k0 + ca);
    async_ld16(boff1[buf], B + (size_t)(n0 + rb1) * ldb + k0 + ca);
  };

  const int nk = K / BK;
  issue(0, 0);                               // prologue
  for (int ki = 0; ki < nk; ++ki) {
    const int cur = ki & 1;
    if (ki + 1 < nk) { issue((ki + 1) * BK, cur ^ 1); wait_async3(); }
    else             { wait_async0(); }
    __syncthreads();                         // tile ki fully in LDS for all waves

    const int bKo = (lane >> 4) * 8;
    v16bf af = frag_ld(&As[cur][wm * 16 + (lane & 15)][bKo]);
#pragma unroll
    for (int s = 0; s < 4; ++s) {
      v16bf bfr = frag_ld(&Bs[cur][wn * 64 + s * 16 + (lane & 15)][bKo]);
      acc[s] = __builtin_amdgcn_wmma_f32_16x16x32_bf16(
          false, af, false, bfr, (short)0, acc[s], false, false);
    }
    __syncthreads();                         // safe to refill this buffer at ki+2
  }

  // D layout: lane L -> col L%16, rows 8*(L/16)+r (vgpr r)
  const int colL = lane & 15, rbase = (lane >> 4) * 8;
#pragma unroll
  for (int s = 0; s < 4; ++s) {
    const int nglob = n0 + wn * 64 + s * 16 + colL;
#pragma unroll
    for (int r = 0; r < 8; ++r) {
      const int mglob = m0 + wm * 16 + rbase + r;
      float v = acc[s][r];
      if constexpr (EPI == 0) {
        ((float*)C)[(size_t)mglob * ldc + nglob] = v;
      } else if constexpr (EPI == 1) {
        ((bf16*)C)[(size_t)mglob * ldc + nglob] = (bf16)siluf(v);
      } else if constexpr (EPI == 2) {
        ((bf16*)C)[(size_t)mglob * ldc + nglob] = (bf16)sigf(v);
      } else {
        float gs = (float)gsig[(size_t)mglob * ldc + nglob];
        ((float*)C)[(size_t)mglob * ldc + nglob] = v * gs;
      }
    }
  }
}

// ---------------------------------------------------------------------------
// Fused gate/up dual GEMM: act = silu(hn@G^T) * (hn@U^T) -> bf16 (ld=DFFNP)
// Block tile 64x64, wave tile 16x32; G/U pre-converted bf16 (1664x1024).
// Same double-buffered async pipeline (3 copies per wave per tile).
// ---------------------------------------------------------------------------
__global__ __launch_bounds__(256) void k_gemm_gu(
    const bf16* __restrict__ A,
    const bf16* __restrict__ G, const bf16* __restrict__ U,
    bf16* __restrict__ act,
    const float* __restrict__ gate, int expert)
{
  __shared__ __align__(16) bf16 As[2][64][BKP];
  __shared__ __align__(16) bf16 Gs[2][64][BKP];
  __shared__ __align__(16) bf16 Us[2][64][BKP];
  __shared__ int s_flag;
  const int tid = threadIdx.x;
  const int m0 = blockIdx.x * 64, n0 = blockIdx.y * 64;

  if (tid == 0) s_flag = 0;
  __syncthreads();
  if (tid < 64 && gate[(size_t)(m0 + tid) * EE + expert] != 0.f) s_flag = 1;
  __syncthreads();
  if (!s_flag) return;

  const int lane = tid & 31, wave = tid >> 5;
  const int wm = wave >> 1, wn = wave & 1;
  v8f accg[2] = {}, accu[2] = {};

  const int ra = tid >> 2, ca = (tid & 3) * 8;  // one 16B granule per 64x32 tile
  const unsigned aoff[2] = { lds_off(&As[0][ra][ca]), lds_off(&As[1][ra][ca]) };
  const unsigned goff[2] = { lds_off(&Gs[0][ra][ca]), lds_off(&Gs[1][ra][ca]) };
  const unsigned uoff[2] = { lds_off(&Us[0][ra][ca]), lds_off(&Us[1][ra][ca]) };

  auto issue = [&](int k0, int buf) {
    async_ld16(aoff[buf], A + (size_t)(m0 + ra) * DD + k0 + ca);
    async_ld16(goff[buf], G + (size_t)(n0 + ra) * DD + k0 + ca);
    async_ld16(uoff[buf], U + (size_t)(n0 + ra) * DD + k0 + ca);
  };

  const int nk = DD / BK;
  issue(0, 0);
  for (int ki = 0; ki < nk; ++ki) {
    const int cur = ki & 1;
    if (ki + 1 < nk) { issue((ki + 1) * BK, cur ^ 1); wait_async3(); }
    else             { wait_async0(); }
    __syncthreads();

    const int bKo = (lane >> 4) * 8;
    v16bf af = frag_ld(&As[cur][wm * 16 + (lane & 15)][bKo]);
#pragma unroll
    for (int s = 0; s < 2; ++s) {
      v16bf gf = frag_ld(&Gs[cur][wn * 32 + s * 16 + (lane & 15)][bKo]);
      accg[s] = __builtin_amdgcn_wmma_f32_16x16x32_bf16(
          false, af, false, gf, (short)0, accg[s], false, false);
      v16bf uf = frag_ld(&Us[cur][wn * 32 + s * 16 + (lane & 15)][bKo]);
      accu[s] = __builtin_amdgcn_wmma_f32_16x16x32_bf16(
          false, af, false, uf, (short)0, accu[s], false, false);
    }
    __syncthreads();
  }

  const int colL = lane & 15, rbase = (lane >> 4) * 8;
#pragma unroll
  for (int s = 0; s < 2; ++s) {
    const int nglob = n0 + wn * 32 + s * 16 + colL;
#pragma unroll
    for (int r = 0; r < 8; ++r) {
      const int mglob = m0 + wm * 16 + rbase + r;
      act[(size_t)mglob * DFFNP + nglob] = (bf16)(siluf(accg[s][r]) * accu[s][r]);
    }
  }
}

// ---------------------------------------------------------------------------
// Per-(expert,token) H coefficients: sigmoids + 4x4 sinkhorn (6 iters).
// Hc layout: (e*NTOK+token)*24: [0..3]=Hpre, [4..7]=Hpost, [8..23]=Hres
// ---------------------------------------------------------------------------
__global__ __launch_bounds__(256) void k_hcoef(
    const float* __restrict__ hlog, const float* __restrict__ gate,
    const float* __restrict__ ap, const float* __restrict__ ao, const float* __restrict__ ar,
    const float* __restrict__ bp, const float* __restrict__ bo, const float* __restrict__ br,
    float* __restrict__ Hc)
{
  const int idx = blockIdx.x * 256 + threadIdx.x;   // EE*NTOK
  const int e = idx >> 12, token = idx & (NTOK - 1);
  if (gate[(size_t)token * EE + e] == 0.f) return;  // contribution exactly 0
  const float* l = hlog + (size_t)token * (EE * 24) + e * 24;
  float* o = Hc + ((size_t)e * NTOK + token) * 24;
  const float a0 = ap[e], a1 = ao[e], a2 = ar[e];
#pragma unroll
  for (int j = 0; j < 4; ++j) o[j]     = sigf(a0 * l[j]     + bp[e * 4 + j]);
#pragma unroll
  for (int j = 0; j < 4; ++j) o[4 + j] = 2.f * sigf(a1 * l[4 + j] + bo[e * 4 + j]);
  float M[16];
#pragma unroll
  for (int k = 0; k < 16; ++k) M[k] = __expf(a2 * l[8 + k] + br[e * 16 + k]);
#pragma unroll
  for (int it = 0; it < 6; ++it) {
#pragma unroll
    for (int i = 0; i < 4; ++i) {
      float rs = M[i*4] + M[i*4+1] + M[i*4+2] + M[i*4+3];
      float inv = 1.f / rs;
      M[i*4] *= inv; M[i*4+1] *= inv; M[i*4+2] *= inv; M[i*4+3] *= inv;
    }
#pragma unroll
    for (int j = 0; j < 4; ++j) {
      float cs = M[j] + M[4+j] + M[8+j] + M[12+j];
      float inv = 1.f / cs;
      M[j] *= inv; M[4+j] *= inv; M[8+j] *= inv; M[12+j] *= inv;
    }
  }
#pragma unroll
  for (int k = 0; k < 16; ++k) o[8 + k] = M[k];
}

// ---------------------------------------------------------------------------
// Per expert: h = sum_n Hpre[n]*stream ; hn = rms(h)*sw_norm -> bf16
// Writes zeros for unrouted tokens so downstream GEMM tiles stay finite.
// ---------------------------------------------------------------------------
__global__ __launch_bounds__(256) void k_h_hn(
    const float* __restrict__ strm, const float* __restrict__ Hc,
    const float* __restrict__ gate, const float* __restrict__ snw,
    bf16* __restrict__ hn, int e)
{
  const int token = blockIdx.x;
  const int b = token >> 11, t = token & (TT - 1);
  const int tid = threadIdx.x, lane = tid & 31, wave = tid >> 5;
  const float g = gate[(size_t)token * EE + e];
  if (g == 0.f) {
#pragma unroll
    for (int q = 0; q < 4; ++q) hn[(size_t)token * DD + tid + q * 256] = (bf16)0.f;
    return;
  }
  const float* hc = Hc + ((size_t)e * NTOK + token) * 24;
  float hp[NN];
#pragma unroll
  for (int n = 0; n < NN; ++n) hp[n] = hc[n];
  const size_t sb = ((size_t)(b * NN) * TT + t) * DD;
  float h[4], ssq = 0.f;
#pragma unroll
  for (int q = 0; q < 4; ++q) {
    const int dd = tid + q * 256;
    float a = 0.f;
#pragma unroll
    for (int n = 0; n < NN; ++n) a += hp[n] * strm[sb + (size_t)n * TT * DD + dd];
    h[q] = a; ssq += a * a;
  }
#pragma unroll
  for (int off = 16; off > 0; off >>= 1) ssq += __shfl_xor(ssq, off, 32);
  __shared__ float sh[8];
  __shared__ float s_scale;
  if (lane == 0) sh[wave] = ssq;
  __syncthreads();
  if (tid == 0) {
    float a = 0.f;
    for (int w = 0; w < 8; ++w) a += sh[w];
    s_scale = rsqrtf(a * (1.0f / DD) + EPSF);
  }
  __syncthreads();
  const float sc = s_scale;
#pragma unroll
  for (int q = 0; q < 4; ++q) {
    const int dd = tid + q * 256;
    hn[(size_t)token * DD + dd] = (bf16)(h[q] * sc * snw[(size_t)e * DD + dd]);
  }
}

// ---------------------------------------------------------------------------
// acc += g * (H_res @ stream + H_post (x) out)
// ---------------------------------------------------------------------------
__global__ __launch_bounds__(256) void k_accum(
    const float* __restrict__ strm, const float* __restrict__ Hc,
    const float* __restrict__ gate, const float* __restrict__ outb,
    float* __restrict__ dst, int e)
{
  const int token = blockIdx.x;
  const int b = token >> 11, t = token & (TT - 1);
  const int tid = threadIdx.x;
  const float g = gate[(size_t)token * EE + e];
  if (g == 0.f) return;
  const float* hc = Hc + ((size_t)e * NTOK + token) * 24;
  float hpost[NN], hres[16];
#pragma unroll
  for (int n = 0; n < NN; ++n) hpost[n] = hc[4 + n];
#pragma unroll
  for (int k = 0; k < 16; ++k) hres[k] = hc[8 + k];
  const size_t sb = ((size_t)(b * NN) * TT + t) * DD;
#pragma unroll
  for (int q = 0; q < 4; ++q) {
    const int dd = tid + q * 256;
    const float o = outb[(size_t)token * DD + dd];
    float sv[NN];
#pragma unroll
    for (int j = 0; j < NN; ++j) sv[j] = strm[sb + (size_t)j * TT * DD + dd];
#pragma unroll
    for (int i = 0; i < NN; ++i) {
      float r = hpost[i] * o;
#pragma unroll
      for (int j = 0; j < NN; ++j) r += hres[i * 4 + j] * sv[j];
      dst[sb + (size_t)i * TT * DD + dd] += g * r;
    }
  }
}

// ---------------------------------------------------------------------------
extern "C" void kernel_launch(void* const* d_in, const int* in_sizes, int n_in,
                              void* d_out, int out_size, void* d_ws, size_t ws_size,
                              hipStream_t stream)
{
  const float* strm     = (const float*)d_in[0];
  const float* router_w = (const float*)d_in[1];
  const float* mhcw     = (const float*)d_in[2];
  const float* pre_w    = (const float*)d_in[3];
  const float* post_w   = (const float*)d_in[4];
  const float* res_w    = (const float*)d_in[5];
  const float* b_pre    = (const float*)d_in[6];
  const float* b_post   = (const float*)d_in[7];
  const float* b_res    = (const float*)d_in[8];
  const float* a_pre    = (const float*)d_in[9];
  const float* a_post   = (const float*)d_in[10];
  const float* a_res    = (const float*)d_in[11];
  const float* snw      = (const float*)d_in[12];
  const float* wd       = (const float*)d_in[13];
  const float* wu       = (const float*)d_in[14];
  const float* gw       = (const float*)d_in[15];
  const float* uw       = (const float*)d_in[16];
  const float* dw       = (const float*)d_in[17];
  (void)in_sizes; (void)n_in; (void)out_size; (void)ws_size;

  const size_t STREAM_ELEMS = (size_t)2 * NN * TT * DD;  // 16,777,216
  float* out_stream = (float*)d_out;
  float* gate = out_stream + STREAM_ELEMS;               // (B,T,E) fp32

  char* w = (char*)d_ws;
  bf16*  base = (bf16*)w;  w += (size_t)NTOK * NDF    * sizeof(bf16);   // 32 MB
  bf16*  WH   = (bf16*)w;  w += (size_t)EE * 24 * NDF * sizeof(bf16);   //  3 MB
  float* hlog = (float*)w; w += (size_t)NTOK * EE * 24 * sizeof(float); //  6 MB
  float* Hc   = (float*)w; w += (size_t)EE * NTOK * 24 * sizeof(float); //  6 MB
  bf16*  hn   = (bf16*)w;  w += (size_t)NTOK * DD     * sizeof(bf16);   //  8 MB
  bf16*  t1   = (bf16*)w;  w += (size_t)NTOK * DD     * sizeof(bf16);   //  8 MB
  bf16*  gsig = (bf16*)w;  w += (size_t)NTOK * DD     * sizeof(bf16);   //  8 MB
  bf16*  act  = (bf16*)w;  w += (size_t)NTOK * DFFNP  * sizeof(bf16);   // 13 MB
  float* outb = (float*)w; w += (size_t)NTOK * DD     * sizeof(float);  // 16 MB
  bf16*  wdb  = (bf16*)w;  w += (size_t)DD * DD       * sizeof(bf16);   //  2 MB
  bf16*  wub  = (bf16*)w;  w += (size_t)DD * DD       * sizeof(bf16);   //  2 MB
  bf16*  gwb  = (bf16*)w;  w += (size_t)DFFNP * DD    * sizeof(bf16);   //  3.25 MB
  bf16*  uwb  = (bf16*)w;  w += (size_t)DFFNP * DD    * sizeof(bf16);   //  3.25 MB
  bf16*  dwb  = (bf16*)w;  w += (size_t)DD * DFFNP    * sizeof(bf16);   //  3.25 MB

  hipMemsetAsync(out_stream, 0, STREAM_ELEMS * sizeof(float), stream);

  k_base_gate<<<NTOK, 256, 0, stream>>>(strm, router_w, base, gate);
  k_fold_wh<<<EE * 24, 256, 0, stream>>>(pre_w, post_w, res_w, mhcw, WH);
  // all H logits in one WMMA GEMM: (4096 tokens) x (384 features) x (4096 K)
  k_gemm<0><<<dim3(NTOK / 64, (EE * 24) / 128), 256, 0, stream>>>(
      base, NDF, WH, NDF, hlog, EE * 24, nullptr, nullptr, 0, NDF);
  k_hcoef<<<(EE * NTOK) / 256, 256, 0, stream>>>(
      hlog, gate, a_pre, a_post, a_res, b_pre, b_post, b_res, Hc);

  const int grpSq = (DD * DD) / (4 * 256);          // 1024 blocks
  const int grpGU = (DFFNP * DD) / (4 * 256);       // 1664 blocks
  for (int e = 0; e < EE; ++e) {
    // one-shot fp32->bf16 weight conversion for this expert (padded, guard-free GEMMs)
    k_cvtw<<<grpSq, 256, 0, stream>>>(wd + (size_t)e * DD * DD,     wdb, DD,   DD,   DD,    DD);
    k_cvtw<<<grpSq, 256, 0, stream>>>(wu + (size_t)e * DD * DD,     wub, DD,   DD,   DD,    DD);
    k_cvtw<<<grpGU, 256, 0, stream>>>(gw + (size_t)e * DFFN * DD,   gwb, DFFN, DD,   DFFNP, DD);
    k_cvtw<<<grpGU, 256, 0, stream>>>(uw + (size_t)e * DFFN * DD,   uwb, DFFN, DD,   DFFNP, DD);
    k_cvtw<<<grpGU, 256, 0, stream>>>(dw + (size_t)e * DD * DFFN,   dwb, DD,   DFFN, DD,    DFFNP);

    k_h_hn<<<NTOK, 256, 0, stream>>>(strm, Hc, gate, snw, hn, e);
    k_gemm<1><<<dim3(NTOK / 64, DD / 128), 256, 0, stream>>>(       // silu(hn@wd^T)
        hn, DD, wdb, DD, t1, DD, nullptr, gate, e, DD);
    k_gemm<2><<<dim3(NTOK / 64, DD / 128), 256, 0, stream>>>(       // sigmoid(t1@wu^T)
        t1, DD, wub, DD, gsig, DD, nullptr, gate, e, DD);
    k_gemm_gu<<<dim3(NTOK / 64, DFFNP / 64), 256, 0, stream>>>(     // silu(g)*u
        hn, gwb, uwb, act, gate, e);
    k_gemm<3><<<dim3(NTOK / 64, DD / 128), 256, 0, stream>>>(       // (act@dw^T)*gsig
        act, DFFNP, dwb, DFFNP, outb, DD, gsig, gate, e, DFFNP);
    k_accum<<<NTOK, 256, 0, stream>>>(strm, Hc, gate, outb, out_stream, e);
  }
}
